// SpatialTransformer_83743272337539
// MI455X (gfx1250) — compile-verified, matrix operand order checked by reading, not run on hardware
//
#include <hip/hip_runtime.h>
#include <hip/hip_bf16.h>
#include <math.h>

typedef unsigned short u16;
typedef unsigned int   u32;

typedef __attribute__((ext_vector_type(16))) __bf16 v16bf;
typedef __attribute__((ext_vector_type(8)))  float  v8f;

#define LSPATIAL 4096   // H*W
#define CCH      512
#define NHEAD    8
#define NPTS     8

// ---------------- helpers ----------------
__device__ inline u16 f2bf(float f) {
    u32 u = __builtin_bit_cast(u32, f);
    u32 r = u + 0x7FFFu + ((u >> 16) & 1u);
    return (u16)(r >> 16);
}

// WMMA fragment load from a 16-row group in LDS (row-major, `stride` u16 elems).
// Per ISA 16-bit A 16x32 layout: lane<16 -> K {0..7,16..23}; lane>=16 -> K {8..15,24..31}.
__device__ inline v16bf ld_frag(const u16* base, int stride, int lane) {
    int r = lane & 15, hi = lane >> 4;
    const u16* p = base + r * stride;
    union { uint4 u[2]; v16bf v; } cv;
    cv.u[0] = *(const uint4*)(p + hi * 8);
    cv.u[1] = *(const uint4*)(p + 16 + hi * 8);
    return cv.v;
}

__device__ inline v8f wmma_bf(v16bf a, v16bf b, v8f c) {
    return __builtin_amdgcn_wmma_f32_16x16x32_bf16(false, a, false, b, (short)0, c, false, false);
}

__device__ inline float gelu_exact(float x) {
    return 0.5f * x * (1.0f + erff(x * 0.70710678118654752f));
}

// ---------------- weight convert: f32 [K][N] -> bf16 [N][K] ----------------
__global__ void wconv_kernel(const float* __restrict__ w, u16* __restrict__ wt, int K, int N) {
    __shared__ float tile[32][33];
    int tx = threadIdx.x, ty = threadIdx.y;
    int n0 = blockIdx.x * 32, k0 = blockIdx.y * 32;
#pragma unroll
    for (int i = 0; i < 4; ++i)
        tile[ty + 8*i][tx] = w[(size_t)(k0 + ty + 8*i) * N + n0 + tx];
    __syncthreads();
#pragma unroll
    for (int i = 0; i < 4; ++i)
        wt[(size_t)(n0 + ty + 8*i) * K + k0 + tx] = f2bf(tile[tx][ty + 8*i]);
}

// ---------------- groupnorm stats: one block per (b,g) ----------------
__global__ __launch_bounds__(256) void gn_stats_kernel(const float* __restrict__ x,
                                                       float* __restrict__ stats) {
    int bg = blockIdx.x, tid = threadIdx.x;
    const float4* p = (const float4*)(x + (size_t)bg * 64 * LSPATIAL);
    float s = 0.f, q = 0.f;
    for (int i = tid; i < 65536; i += 256) {
        float4 v = p[i];
        s += v.x + v.y + v.z + v.w;
        q += v.x*v.x + v.y*v.y + v.z*v.z + v.w*v.w;
    }
    __shared__ float s1[256], s2[256];
    s1[tid] = s; s2[tid] = q;
    __syncthreads();
    for (int st = 128; st > 0; st >>= 1) {
        if (tid < st) { s1[tid] += s1[tid + st]; s2[tid] += s2[tid + st]; }
        __syncthreads();
    }
    if (tid == 0) {
        float mean = s1[0] * (1.0f / 262144.0f);
        float var  = s2[0] * (1.0f / 262144.0f) - mean * mean;
        stats[bg * 2]     = mean;
        stats[bg * 2 + 1] = rsqrtf(var + 1e-5f);
    }
}

// ---------------- groupnorm apply + transpose (B,C,L)->(B,L,C) bf16 ----------------
__global__ void gn_apply_kernel(const float* __restrict__ x, const float* __restrict__ gg,
                                const float* __restrict__ gb, const float* __restrict__ stats,
                                u16* __restrict__ outB) {
    __shared__ float tile[32][33];
    int tx = threadIdx.x, ty = threadIdx.y;
    int l0 = blockIdx.x * 32, c0 = blockIdx.y * 32, b = blockIdx.z;
#pragma unroll
    for (int i = 0; i < 4; ++i) {
        int c = c0 + ty + 8*i;
        tile[ty + 8*i][tx] = x[((size_t)b * CCH + c) * LSPATIAL + l0 + tx];
    }
    __syncthreads();
#pragma unroll
    for (int i = 0; i < 4; ++i) {
        int l = l0 + ty + 8*i;
        int c = c0 + tx;
        int g = c >> 6;
        float mean = stats[(b * 8 + g) * 2];
        float rstd = stats[(b * 8 + g) * 2 + 1];
        float sc = gg[c] * rstd;
        float sh = gb[c] - mean * sc;
        outB[((size_t)b * LSPATIAL + l) * CCH + c] = f2bf(tile[tx][ty + 8*i] * sc + sh);
    }
}

// ---------------- layernorm over C=512 (block per row) ----------------
__global__ __launch_bounds__(256) void layernorm_kernel(const float* __restrict__ in,
                                                        const float* __restrict__ g,
                                                        const float* __restrict__ b,
                                                        float* __restrict__ outF,
                                                        u16* __restrict__ outB) {
    int row = blockIdx.x, tid = threadIdx.x;
    const float* p = in + (size_t)row * CCH;
    float x0 = p[tid], x1 = p[tid + 256];
    __shared__ float s1[256], s2[256];
    s1[tid] = x0 + x1; s2[tid] = x0*x0 + x1*x1;
    __syncthreads();
    for (int st = 128; st > 0; st >>= 1) {
        if (tid < st) { s1[tid] += s1[tid + st]; s2[tid] += s2[tid + st]; }
        __syncthreads();
    }
    float mean = s1[0] * (1.0f / 512.0f);
    float var  = s2[0] * (1.0f / 512.0f) - mean * mean;
    float rstd = rsqrtf(var + 1e-5f);
    float y0 = (x0 - mean) * rstd * g[tid]       + b[tid];
    float y1 = (x1 - mean) * rstd * g[tid + 256] + b[tid + 256];
    size_t o = (size_t)row * CCH + tid;
    if (outF) { outF[o] = y0; outF[o + 256] = y1; }
    if (outB) { outB[o] = f2bf(y0); outB[o + 256] = f2bf(y1); }
}

// ---------------- softmax over NP=8, thread per (m,h) ----------------
__global__ void softmax8_kernel(float* __restrict__ aw) {
    int gid = blockIdx.x * blockDim.x + threadIdx.x;   // < M*NH
    float* p = aw + (size_t)gid * 8;                   // m*64 + h*8 == gid*8
    float v[8], mx = -1e30f;
#pragma unroll
    for (int i = 0; i < 8; ++i) { v[i] = p[i]; mx = fmaxf(mx, v[i]); }
    float s = 0.f;
#pragma unroll
    for (int i = 0; i < 8; ++i) { v[i] = __expf(v[i] - mx); s += v[i]; }
    float inv = 1.0f / s;
#pragma unroll
    for (int i = 0; i < 8; ++i) p[i] = v[i] * inv;
}

// ---------------- MSDA sampler: wave per (b,l,h), 2 dims per lane ----------------
__global__ __launch_bounds__(256) void msda_kernel(const float* __restrict__ v2,
                                                   const float* __restrict__ soff,
                                                   const float* __restrict__ aw,
                                                   u16* __restrict__ outB) {
    int gtid = blockIdx.x * 256 + threadIdx.x;
    int wave = gtid >> 5;           // [0, M*NH)
    int lane = gtid & 31;
    int h  = wave & 7;
    int ml = wave >> 3;             // b*L + l
    int b  = ml >> 12;
    const float py_bias = 0.5f * (float)(LSPATIAL - 1);   // Hl=L, Wl=1

    float acc0 = 0.f, acc1 = 0.f;
#pragma unroll
    for (int p = 0; p < NPTS; ++p) {
        int pi = h * NPTS + p;
        float offx = soff[(size_t)ml * 128 + pi * 2 + 0];
        float offy = soff[(size_t)ml * 128 + pi * 2 + 1];
        float a = aw[(size_t)ml * 64 + pi];
        float px = offx;                 // + 0.5*(Wl-1) == 0
        float py = offy + py_bias;
        float x0f = floorf(px), y0f = floorf(py);
        float lx = px - x0f, ly = py - y0f;
        int x0 = (int)x0f, y0 = (int)y0f;
        float s0 = 0.f, s1 = 0.f;
#pragma unroll
        for (int cy = 0; cy < 2; ++cy) {
#pragma unroll
            for (int cx = 0; cx < 2; ++cx) {
                int xi = x0 + cx, yi = y0 + cy;
                float wx = cx ? lx : (1.f - lx);
                float wy = cy ? ly : (1.f - ly);
                float wv = wx * wy;
                bool valid = (xi >= 0) && (xi < 1) && (yi >= 0) && (yi < LSPATIAL);
                if (valid && wv != 0.f) {
                    size_t base = ((size_t)(b << 12) + (size_t)yi) * CCH + h * 64;
                    s0 += v2[base + lane]      * wv;
                    s1 += v2[base + 32 + lane] * wv;
                }
            }
        }
        acc0 += a * s0;
        acc1 += a * s1;
    }
    size_t ob = (size_t)ml * CCH + h * 64;
    outB[ob + lane]      = f2bf(acc0);
    outB[ob + 32 + lane] = f2bf(acc1);
}

// ======================================================================
// Wide WMMA GEMM: block tile 128x128, 8 waves, each wave 32x64 (2x4 WMMA).
// C[M,N] = A[M,K] * Bt[N,K]^T  (N multiple of 128)
// mode 0: outF = acc+bias
// mode 1: outB = bf16(acc+bias)
// mode 2: outB = bf16(acc+bias+add)
// mode 3: outF = acc+bias+add
// mode 4: outF[(b*512+col)*4096+l] = acc+bias+xin[same]   (N==512), float4 stores
// ======================================================================
__global__ __launch_bounds__(256) void gemm_bf16_wide(
    const u16* __restrict__ A, const u16* __restrict__ Bt,
    const float* __restrict__ bias, const float* __restrict__ add,
    float* __restrict__ outF, u16* __restrict__ outB,
    const float* __restrict__ xin, int M, int N, int K, int mode) {
    constexpr int BM = 128, BN = 128, BK = 32, STR = BK + 8;
    __shared__ u16 As[BM * STR];
    __shared__ u16 Bs[BN * STR];
    int tid = threadIdx.x;
    int lane = tid & 31, wid = tid >> 5;
    int wm = wid & 3, wn = wid >> 2;            // wave tile: rows wm*32, cols wn*64
    int rowBase = blockIdx.y * BM;
    int colBase = blockIdx.x * BN;
    v8f acc[2][4] = {};
    for (int k0 = 0; k0 < K; k0 += BK) {
        __syncthreads();
#pragma unroll
        for (int it = 0; it < 2; ++it) {
            int c = tid + 256 * it;
            int row = c >> 2, kk = (c & 3) << 3;
            *(uint4*)&As[row * STR + kk] =
                *(const uint4*)&A[(size_t)(rowBase + row) * K + k0 + kk];
            *(uint4*)&Bs[row * STR + kk] =
                *(const uint4*)&Bt[(size_t)(colBase + row) * K + k0 + kk];
        }
        __syncthreads();
        v16bf a0 = ld_frag(&As[(wm * 32 +  0) * STR], STR, lane);
        v16bf a1 = ld_frag(&As[(wm * 32 + 16) * STR], STR, lane);
#pragma unroll
        for (int j = 0; j < 4; ++j) {
            v16bf bj = ld_frag(&Bs[(wn * 64 + j * 16) * STR], STR, lane);
            acc[0][j] = wmma_bf(a0, bj, acc[0][j]);
            acc[1][j] = wmma_bf(a1, bj, acc[1][j]);
        }
    }
    int ln = lane & 15, hb = (lane >> 4) * 8;
#pragma unroll
    for (int i = 0; i < 2; ++i)
#pragma unroll
        for (int j = 0; j < 4; ++j) {
            int col = colBase + wn * 64 + j * 16 + ln;
            int m0  = rowBase + wm * 32 + i * 16 + hb;
            float bv = bias[col];
            if (mode == 4) {
                // transposed + residual x; 8 consecutive l -> two float4 segments
                int bb = m0 >> 12, l = m0 & 4095;
                size_t o = ((size_t)(bb * CCH + col) << 12) + l;
                float4 x0 = *(const float4*)&xin[o];
                float4 x1 = *(const float4*)&xin[o + 4];
                float4 r0, r1;
                r0.x = acc[i][j][0] + bv + x0.x;
                r0.y = acc[i][j][1] + bv + x0.y;
                r0.z = acc[i][j][2] + bv + x0.z;
                r0.w = acc[i][j][3] + bv + x0.w;
                r1.x = acc[i][j][4] + bv + x1.x;
                r1.y = acc[i][j][5] + bv + x1.y;
                r1.z = acc[i][j][6] + bv + x1.z;
                r1.w = acc[i][j][7] + bv + x1.w;
                *(float4*)&outF[o]     = r0;
                *(float4*)&outF[o + 4] = r1;
            } else {
#pragma unroll
                for (int r = 0; r < 8; ++r) {
                    int m = m0 + r;
                    float v = acc[i][j][r] + bv;
                    if (mode == 0) {
                        outF[(size_t)m * N + col] = v;
                    } else if (mode == 1) {
                        outB[(size_t)m * N + col] = f2bf(v);
                    } else if (mode == 2) {
                        outB[(size_t)m * N + col] = f2bf(v + add[(size_t)m * N + col]);
                    } else {  // mode 3
                        outF[(size_t)m * N + col] = v + add[(size_t)m * N + col];
                    }
                }
            }
        }
}

// ---------------- narrow WMMA GEMM (BN=64), used for attw (N=64) ----------------
__global__ __launch_bounds__(256) void gemm_bf16_narrow(
    const u16* __restrict__ A, const u16* __restrict__ Bt,
    const float* __restrict__ bias, float* __restrict__ outF,
    int M, int N, int K) {
    constexpr int BM = 128, BN = 64, BK = 32, STR = BK + 8;
    __shared__ u16 As[BM * STR];
    __shared__ u16 Bs[BN * STR];
    int tid = threadIdx.x;
    int lane = tid & 31, wid = tid >> 5;
    int wm = wid & 3, wn = wid >> 2;
    int rowBase = blockIdx.y * BM;
    int colBase = blockIdx.x * BN;
    v8f acc[2][2] = {};
    for (int k0 = 0; k0 < K; k0 += BK) {
        __syncthreads();
#pragma unroll
        for (int it = 0; it < 2; ++it) {
            int c = tid + 256 * it;
            int row = c >> 2, kk = (c & 3) << 3;
            *(uint4*)&As[row * STR + kk] =
                *(const uint4*)&A[(size_t)(rowBase + row) * K + k0 + kk];
        }
        {
            int row = tid >> 2, kk = (tid & 3) << 3;
            *(uint4*)&Bs[row * STR + kk] =
                *(const uint4*)&Bt[(size_t)(colBase + row) * K + k0 + kk];
        }
        __syncthreads();
        v16bf a0 = ld_frag(&As[(wm * 32 +  0) * STR], STR, lane);
        v16bf a1 = ld_frag(&As[(wm * 32 + 16) * STR], STR, lane);
        v16bf b0 = ld_frag(&Bs[(wn * 32 +  0) * STR], STR, lane);
        v16bf b1 = ld_frag(&Bs[(wn * 32 + 16) * STR], STR, lane);
        acc[0][0] = wmma_bf(a0, b0, acc[0][0]);
        acc[0][1] = wmma_bf(a0, b1, acc[0][1]);
        acc[1][0] = wmma_bf(a1, b0, acc[1][0]);
        acc[1][1] = wmma_bf(a1, b1, acc[1][1]);
    }
    int ln = lane & 15, hb = (lane >> 4) * 8;
#pragma unroll
    for (int i = 0; i < 2; ++i)
#pragma unroll
        for (int j = 0; j < 2; ++j) {
            int col = colBase + wn * 32 + j * 16 + ln;
            int m0  = rowBase + wm * 32 + i * 16 + hb;
            float bv = bias[col];
#pragma unroll
            for (int r = 0; r < 8; ++r)
                outF[(size_t)(m0 + r) * N + col] = acc[i][j][r] + bv;
        }
}

// ---------------- geglu-fused WMMA GEMM: out = (A*Wa+ba) * gelu(A*Wg+bg) ----------------
// Wt is [4096][K]; 'a' uses cols [0,2048), gate uses cols [2048,4096).
__global__ __launch_bounds__(256) void gemm_geglu_kernel(
    const u16* __restrict__ A, const u16* __restrict__ Wt,
    const float* __restrict__ bias, u16* __restrict__ outB, int M, int K) {
    constexpr int BM = 128, BN = 64, BK = 32, STR = BK + 8, HALF = 2048;
    __shared__ u16 As[BM * STR];
    __shared__ u16 Ba[BN * STR];
    __shared__ u16 Bg[BN * STR];
    int tid = threadIdx.x;
    int lane = tid & 31, wid = tid >> 5;
    int wm = wid & 3, wn = wid >> 2;
    int rowBase = blockIdx.y * BM;
    int colBase = blockIdx.x * BN;
    v8f accA[2][2] = {};
    v8f accG[2][2] = {};
    for (int k0 = 0; k0 < K; k0 += BK) {
        __syncthreads();
#pragma unroll
        for (int it = 0; it < 2; ++it) {
            int c = tid + 256 * it;
            int row = c >> 2, kk = (c & 3) << 3;
            *(uint4*)&As[row * STR + kk] =
                *(const uint4*)&A[(size_t)(rowBase + row) * K + k0 + kk];
        }
        {
            int row = tid >> 2, kk = (tid & 3) << 3;
            *(uint4*)&Ba[row * STR + kk] =
                *(const uint4*)&Wt[(size_t)(colBase + row) * K + k0 + kk];
            *(uint4*)&Bg[row * STR + kk] =
                *(const uint4*)&Wt[(size_t)(HALF + colBase + row) * K + k0 + kk];
        }
        __syncthreads();
        v16bf a0  = ld_frag(&As[(wm * 32 +  0) * STR], STR, lane);
        v16bf a1  = ld_frag(&As[(wm * 32 + 16) * STR], STR, lane);
        v16bf ba0 = ld_frag(&Ba[(wn * 32 +  0) * STR], STR, lane);
        v16bf ba1 = ld_frag(&Ba[(wn * 32 + 16) * STR], STR, lane);
        v16bf bg0 = ld_frag(&Bg[(wn * 32 +  0) * STR], STR, lane);
        v16bf bg1 = ld_frag(&Bg[(wn * 32 + 16) * STR], STR, lane);
        accA[0][0] = wmma_bf(a0, ba0, accA[0][0]);
        accA[0][1] = wmma_bf(a0, ba1, accA[0][1]);
        accA[1][0] = wmma_bf(a1, ba0, accA[1][0]);
        accA[1][1] = wmma_bf(a1, ba1, accA[1][1]);
        accG[0][0] = wmma_bf(a0, bg0, accG[0][0]);
        accG[0][1] = wmma_bf(a0, bg1, accG[0][1]);
        accG[1][0] = wmma_bf(a1, bg0, accG[1][0]);
        accG[1][1] = wmma_bf(a1, bg1, accG[1][1]);
    }
    int ln = lane & 15, hb = (lane >> 4) * 8;
#pragma unroll
    for (int i = 0; i < 2; ++i)
#pragma unroll
        for (int j = 0; j < 2; ++j) {
            int col = colBase + wn * 32 + j * 16 + ln;
            int m0  = rowBase + wm * 32 + i * 16 + hb;
            float ba = bias[col];
            float bg = bias[HALF + col];
#pragma unroll
            for (int r = 0; r < 8; ++r) {
                int m = m0 + r;
                float va = accA[i][j][r] + ba;
                float vg = accG[i][j][r] + bg;
                outB[(size_t)m * HALF + col] = f2bf(va * gelu_exact(vg));
            }
        }
}

// ---------------- host: launch pipeline ----------------
extern "C" void kernel_launch(void* const* d_in, const int* in_sizes, int n_in,
                              void* d_out, int out_size, void* d_ws, size_t ws_size,
                              hipStream_t stream) {
    (void)in_sizes; (void)n_in; (void)out_size; (void)ws_size;
    const float* x       = (const float*)d_in[0];
    const float* gn_g    = (const float*)d_in[1];
    const float* gn_b    = (const float*)d_in[2];
    const float* pin_w   = (const float*)d_in[3];
    const float* pin_b   = (const float*)d_in[4];
    const float* ln1_g   = (const float*)d_in[5];
    const float* ln1_b   = (const float*)d_in[6];
    const float* vproj_w = (const float*)d_in[7];
    const float* vproj_b = (const float*)d_in[8];
    const float* mvp_w   = (const float*)d_in[9];
    const float* mvp_b   = (const float*)d_in[10];
    const float* soff_w  = (const float*)d_in[11];
    const float* soff_b  = (const float*)d_in[12];
    const float* attw_w  = (const float*)d_in[13];
    const float* attw_b  = (const float*)d_in[14];
    const float* mop_w   = (const float*)d_in[15];
    const float* mop_b   = (const float*)d_in[16];
    const float* dout_w  = (const float*)d_in[17];
    const float* dout_b  = (const float*)d_in[18];
    const float* ln3_g   = (const float*)d_in[19];
    const float* ln3_b   = (const float*)d_in[20];
    const float* geglu_w = (const float*)d_in[21];
    const float* geglu_b = (const float*)d_in[22];
    const float* dense_w = (const float*)d_in[23];
    const float* dense_b = (const float*)d_in[24];
    const float* pout_w  = (const float*)d_in[25];
    const float* pout_b  = (const float*)d_in[26];
    float* out = (float*)d_out;

    const int M = 4 * LSPATIAL;  // 16384

    char* ws = (char*)d_ws;
    size_t off = 0;
    auto alloc = [&](size_t bytes) -> char* {
        off = (off + 255) & ~(size_t)255;
        char* p = ws + off;
        off += bytes;
        return p;
    };

    u16* wt_pin   = (u16*)alloc((size_t)512 * 512 * 2);
    u16* wt_vproj = (u16*)alloc((size_t)512 * 512 * 2);
    u16* wt_mvp   = (u16*)alloc((size_t)512 * 512 * 2);
    u16* wt_soff  = (u16*)alloc((size_t)128 * 512 * 2);
    u16* wt_attw  = (u16*)alloc((size_t)64  * 512 * 2);
    u16* wt_mop   = (u16*)alloc((size_t)512 * 512 * 2);
    u16* wt_dout  = (u16*)alloc((size_t)512 * 512 * 2);
    u16* wt_geglu = (u16*)alloc((size_t)4096 * 512 * 2);
    u16* wt_dense = (u16*)alloc((size_t)512 * 2048 * 2);
    u16* wt_pout  = (u16*)alloc((size_t)512 * 512 * 2);
    float* stats  = (float*)alloc(32 * 2 * sizeof(float));
    u16*   xnb    = (u16*)alloc((size_t)M * 512 * 2);   // later reused as hn_bf16
    float* t0f    = (float*)alloc((size_t)M * 512 * 4);
    float* qf     = (float*)alloc((size_t)M * 512 * 4);
    u16*   qb     = (u16*)alloc((size_t)M * 512 * 2);   // later reused as m1_bf16
    u16*   valb   = (u16*)alloc((size_t)M * 512 * 2);   // later reused as msda_bf16
    float* v2f    = (float*)alloc((size_t)M * 512 * 4); // later reused as t1_f32
    float* soffo  = (float*)alloc((size_t)M * 128 * 4);
    float* awf    = (float*)alloc((size_t)M * 64 * 4);
    u16*   ffinb  = (u16*)alloc((size_t)M * 2048 * 2);
    u16*   t2b    = (u16*)alloc((size_t)M * 512 * 2);

    u16* hnb   = xnb;
    u16* m1b   = qb;
    u16* msdab = valb;
    float* t1f = v2f;

    dim3 tb(32, 8);
    // weight conversion (f32 [K][N] -> bf16 [N][K])
    wconv_kernel<<<dim3(16, 16), tb, 0, stream>>>(pin_w,   wt_pin,   512, 512);
    wconv_kernel<<<dim3(16, 16), tb, 0, stream>>>(vproj_w, wt_vproj, 512, 512);
    wconv_kernel<<<dim3(16, 16), tb, 0, stream>>>(mvp_w,   wt_mvp,   512, 512);
    wconv_kernel<<<dim3(4, 16),  tb, 0, stream>>>(soff_w,  wt_soff,  512, 128);
    wconv_kernel<<<dim3(2, 16),  tb, 0, stream>>>(attw_w,  wt_attw,  512, 64);
    wconv_kernel<<<dim3(16, 16), tb, 0, stream>>>(mop_w,   wt_mop,   512, 512);
    wconv_kernel<<<dim3(16, 16), tb, 0, stream>>>(dout_w,  wt_dout,  512, 512);
    wconv_kernel<<<dim3(128, 16), tb, 0, stream>>>(geglu_w, wt_geglu, 512, 4096);
    wconv_kernel<<<dim3(16, 64), tb, 0, stream>>>(dense_w, wt_dense, 2048, 512);
    wconv_kernel<<<dim3(16, 16), tb, 0, stream>>>(pout_w,  wt_pout,  512, 512);

    // 1) groupnorm + transpose to (B,L,C) bf16
    gn_stats_kernel<<<32, 256, 0, stream>>>(x, stats);
    gn_apply_kernel<<<dim3(128, 16, 4), tb, 0, stream>>>(x, gn_g, gn_b, stats, xnb);

    const int GR = M / 128;  // 128 row-blocks
    // 2) pin: t0 = xn @ pin_w + b  (f32)
    gemm_bf16_wide<<<dim3(4, GR), 256, 0, stream>>>(xnb, wt_pin, pin_b, nullptr,
                                                    t0f, nullptr, nullptr, M, 512, 512, 0);
    // 3) q = LN(t0) -> f32 + bf16
    layernorm_kernel<<<M, 256, 0, stream>>>(t0f, ln1_g, ln1_b, qf, qb);
    // 4) value = q @ vproj + b (bf16)
    gemm_bf16_wide<<<dim3(4, GR), 256, 0, stream>>>(qb, wt_vproj, vproj_b, nullptr,
                                                    nullptr, valb, nullptr, M, 512, 512, 1);
    // 5) v2 = value @ mvp + b (f32, sampled later)
    gemm_bf16_wide<<<dim3(4, GR), 256, 0, stream>>>(valb, wt_mvp, mvp_b, nullptr,
                                                    v2f, nullptr, nullptr, M, 512, 512, 0);
    // 6) sampling offsets and attention weights
    gemm_bf16_wide<<<dim3(1, GR), 256, 0, stream>>>(qb, wt_soff, soff_b, nullptr,
                                                    soffo, nullptr, nullptr, M, 128, 512, 0);
    gemm_bf16_narrow<<<dim3(1, GR), 256, 0, stream>>>(qb, wt_attw, attw_b, awf, M, 64, 512);
    softmax8_kernel<<<(M * NHEAD) / 256, 256, 0, stream>>>(awf);
    // 7) deformable sampling -> msda (bf16)
    msda_kernel<<<(M * NHEAD * 32) / 256, 256, 0, stream>>>(v2f, soffo, awf, msdab);
    // 8) m1 = msda @ mop + b + q (bf16)
    gemm_bf16_wide<<<dim3(4, GR), 256, 0, stream>>>(msdab, wt_mop, mop_b, qf,
                                                    nullptr, m1b, nullptr, M, 512, 512, 2);
    // 9) t1 = m1 @ dout + b + t0 (f32)
    gemm_bf16_wide<<<dim3(4, GR), 256, 0, stream>>>(m1b, wt_dout, dout_b, t0f,
                                                    t1f, nullptr, nullptr, M, 512, 512, 3);
    // 10) hn = LN(t1) (bf16 only)
    layernorm_kernel<<<M, 256, 0, stream>>>(t1f, ln3_g, ln3_b, nullptr, hnb);
    // 11) ffin = (hn@Wa+ba) * gelu(hn@Wg+bg)  (bf16, fused)
    gemm_geglu_kernel<<<dim3(32, GR), 256, 0, stream>>>(hnb, wt_geglu, geglu_b, ffinb, M, 512);
    // 12) t2 = ffin @ dense + b + t1 (bf16)
    gemm_bf16_wide<<<dim3(4, GR), 256, 0, stream>>>(ffinb, wt_dense, dense_b, t1f,
                                                    nullptr, t2b, nullptr, M, 512, 2048, 2);
    // 13) out = transpose(t2 @ pout + b) + x  -> (B,C,H,W)
    gemm_bf16_wide<<<dim3(4, GR), 256, 0, stream>>>(t2b, wt_pout, pout_b, nullptr,
                                                    out, nullptr, x, M, 512, 512, 4);
}
